// HypergraphFeatureEnhancer_13005160972462
// MI455X (gfx1250) — matrix operand order repaired and checked by reading, not compiled, not run on hardware
//
#include <hip/hip_runtime.h>

typedef __attribute__((ext_vector_type(16))) _Float16 v16h;
typedef __attribute__((ext_vector_type(8)))  _Float16 v8h;
typedef __attribute__((ext_vector_type(8)))  float    v8f;
typedef unsigned int v4u __attribute__((ext_vector_type(4)));
typedef int          v4i __attribute__((ext_vector_type(4)));
typedef int          v8i __attribute__((ext_vector_type(8)));

static constexpr int NODES = 4096;   // H*W
static constexpr int CH    = 256;    // C == HID
static constexpr int BATCH = 4;
static constexpr int KTOP  = 8;
static constexpr int ROWS  = BATCH * NODES; // 16384

// ---------------------------------------------------------------------------
// TDM helper: 2D tile (tile_w contiguous elems x tile_h rows, 2-byte elems)
// global -> LDS with DWORD padding appended every row.
//   pad_interval_code: 0=2,1=4,2=8,3=16,4=32,... DWORDs between pads
//   pad_amount_code:   (code+1) DWORDs of pad
// ---------------------------------------------------------------------------
__device__ __forceinline__ void tdm_load_2d_f16(unsigned lds_byte_off,
                                                const void* gaddr,
                                                int tile_w, int tile_h,
                                                long row_stride_elems,
                                                int pad_interval_code,
                                                int pad_amount_code) {
  unsigned long long ga = (unsigned long long)gaddr;
  v4u g0;
  g0[0] = 1u;                                        // count=1, user mode
  g0[1] = lds_byte_off;                              // lds_addr
  g0[2] = (unsigned)(ga & 0xFFFFFFFFu);              // global_addr[31:0]
  g0[3] = (unsigned)((ga >> 32) & 0x01FFFFFFu) | (2u << 30); // addr[56:32] | type=2
  unsigned td0 = 1u << 30, td1 = 1u << 30;           // huge tensor dims: no OOB clip
  unsigned long long st = (unsigned long long)row_stride_elems;
  v8i g1;
  g1[0] = (int)((1u << 16)                            // data_size = 2 bytes
              | (1u << 20)                            // pad_enable
              | ((unsigned)pad_interval_code << 22)
              | ((unsigned)pad_amount_code << 25));
  g1[1] = (int)((td0 & 0xFFFFu) << 16);               // tensor_dim0[15:0]
  g1[2] = (int)(((td0 >> 16) & 0xFFFFu) | ((td1 & 0xFFFFu) << 16));
  g1[3] = (int)(((td1 >> 16) & 0xFFFFu) | ((unsigned)tile_w << 16)); // tile_dim0
  g1[4] = tile_h & 0xFFFF;                            // tile_dim1 (tile_dim2=0)
  g1[5] = (int)(st & 0xFFFFFFFFu);                    // tensor_dim0_stride lo
  g1[6] = (int)((st >> 32) & 0xFFFFu);                // stride hi (dim1_stride=0)
  g1[7] = 0;
  v4i z4 = {0, 0, 0, 0};
#if defined(__clang_major__) && (__clang_major__ >= 23)
  v8i z8 = {0, 0, 0, 0, 0, 0, 0, 0};
  __builtin_amdgcn_tensor_load_to_lds(g0, g1, z4, z4, z8, 0);
#else
  __builtin_amdgcn_tensor_load_to_lds(g0, g1, z4, z4, 0);
#endif
}

// ---------------------------------------------------------------------------
// 1) Transpose + cast: features [B][C][N] f32  ->  X [B*N][C] f16
// ---------------------------------------------------------------------------
__global__ void transpose_cast_kernel(const float* __restrict__ feat,
                                      _Float16* __restrict__ Xh) {
  __shared__ float t[32][33];
  int b  = blockIdx.z;
  int c0 = blockIdx.x * 32;
  int n0 = blockIdx.y * 32;
  int tx = threadIdx.x, ty = threadIdx.y; // 32 x 8
#pragma unroll
  for (int i = 0; i < 4; i++) {
    int c = c0 + ty + i * 8;
    t[ty + i * 8][tx] = feat[((long)b * CH + c) * NODES + (n0 + tx)];
  }
  __syncthreads();
#pragma unroll
  for (int i = 0; i < 4; i++) {
    int n = n0 + ty + i * 8;
    Xh[((long)b * NODES + n) * CH + (c0 + tx)] = (_Float16)t[tx][ty + i * 8];
  }
}

// ---------------------------------------------------------------------------
// 2) Row L2-normalize batch-0 features -> f0n f16 [N][C]
// ---------------------------------------------------------------------------
__global__ void normalize_f0_kernel(const float* __restrict__ feat,
                                    _Float16* __restrict__ f0n) {
  int n = blockIdx.x * blockDim.x + threadIdx.x; // 0..4095
  float ss = 0.f;
  for (int c = 0; c < CH; c++) {
    float v = feat[(long)c * NODES + n];
    ss += v * v;
  }
  float nrm = fmaxf(sqrtf(ss), 1e-12f);
  float ri = 1.0f / nrm;
  for (int c = 0; c < CH; c++)
    f0n[(long)n * CH + c] = (_Float16)(feat[(long)c * NODES + n] * ri);
}

// ---------------------------------------------------------------------------
// Top-k helpers (descending by value, ties -> smaller index, like jax top_k)
// ---------------------------------------------------------------------------
__device__ __forceinline__ bool tk_better(float av, int ai, float bv, int bi) {
  return (av > bv) || (av == bv && ai < bi);
}
__device__ __forceinline__ void tk_insert(float (&tv)[KTOP], int (&ti)[KTOP],
                                          float v, int id) {
  if (!tk_better(v, id, tv[KTOP - 1], ti[KTOP - 1])) return;
  tv[KTOP - 1] = v; ti[KTOP - 1] = id;
#pragma unroll
  for (int p = KTOP - 1; p > 0; p--) {
    if (tk_better(tv[p], ti[p], tv[p - 1], ti[p - 1])) {
      float fv = tv[p]; tv[p] = tv[p - 1]; tv[p - 1] = fv;
      int fi = ti[p]; ti[p] = ti[p - 1]; ti[p - 1] = fi;
    }
  }
}

// ---------------------------------------------------------------------------
// 3) Fused sim = f0n @ f0n^T (f16 WMMA, fp32 accum) + per-row top-8.
//    One block per 16 sim rows; fragments loaded straight from global
//    (contiguous v8h/v16h per-lane loads). The A-row pointer is laundered
//    through an empty asm each iteration so LICM cannot hoist the fragment
//    loads out of the column loop (which previously created a 64-VGPR live
//    range and scratch spills); reloads hit L0 since rows are reused 32x.
// ---------------------------------------------------------------------------
__global__ void sim_topk_kernel(const _Float16* __restrict__ f0n,
                                int* __restrict__ topk) {
  int tid  = threadIdx.x;
  int wave = tid >> 5;
  int lane = tid & 31;
  int hi   = lane >> 4;
  int lm   = lane & 15;
  int rowbase = blockIdx.x * 16;

  __shared__ float Ct[8][16][16];
  __shared__ float mval[16][64];
  __shared__ int   midx[16][64];

  int m = rowbase + lm;
  const v8h* arow_base = (const v8h*)(f0n + (long)m * CH);

  float tv[KTOP]; int ti[KTOP];
#pragma unroll
  for (int i = 0; i < KTOP; i++) { tv[i] = -3.0e38f; ti[i] = 0x7fffffff; }

  // Each wave sweeps 32 of the 256 column tiles.
  for (int ct = wave; ct < 256; ct += 8) {
    int colbase = ct * 16;
    int col = colbase + lm;
    const _Float16* brow = f0n + (long)col * CH;
    const v8h* arow = arow_base;
    asm volatile("" : "+v"(arow));   // opaque: defeat LICM / keep VGPRs low
    v8f acc;
#pragma unroll
    for (int i = 0; i < 8; i++) acc[i] = 0.0f;
#pragma unroll
    for (int ks = 0; ks < 8; ks++) {
      v8h lo = arow[ks * 4 + hi * 2];       // k = ks*32 + hi*16 + [0..7]
      v8h hv = arow[ks * 4 + hi * 2 + 1];   // k = ks*32 + hi*16 + [8..15]
      v16h af = __builtin_shufflevector(lo, hv, 0, 1, 2, 3, 4, 5, 6, 7,
                                        8, 9, 10, 11, 12, 13, 14, 15);
      v16h bf = *(const v16h*)(brow + ks * 32 + hi * 16); // k = ks*32+hi*16+[0..15]
      acc = __builtin_amdgcn_wmma_f32_16x16x32_f16(
          false, af, false, bf, (short)0, acc, false, false);
    }
#pragma unroll
    for (int v = 0; v < 8; v++) Ct[wave][v + 8 * hi][lm] = acc[v];
    __syncthreads();
    if (lane < 16) {
#pragma unroll
      for (int cc = 0; cc < 16; cc++)
        tk_insert(tv, ti, Ct[wave][lane][cc], colbase + cc);
    }
    __syncthreads();
  }

  // Merge 8 per-wave top-8 lists per row.
  if (lane < 16) {
#pragma unroll
    for (int k = 0; k < KTOP; k++) {
      mval[lane][wave * 8 + k] = tv[k];
      midx[lane][wave * 8 + k] = ti[k];
    }
  }
  __syncthreads();
  if (tid < 16) {
    float fv[KTOP]; int fi[KTOP];
#pragma unroll
    for (int i = 0; i < KTOP; i++) { fv[i] = -3.0e38f; fi[i] = 0x7fffffff; }
    for (int j = 0; j < 64; j++) tk_insert(fv, fi, mval[tid][j], midx[tid][j]);
#pragma unroll
    for (int k = 0; k < KTOP; k++)
      topk[(long)(rowbase + tid) * KTOP + k] = fi[k];
  }
}

// ---------------------------------------------------------------------------
// 4) Build dense Hn (column-L1-normalized incidence): Hn[r][e] = 0.125
// ---------------------------------------------------------------------------
__global__ void zero_u32_kernel(unsigned* __restrict__ p, long n) {
  long i = (long)blockIdx.x * blockDim.x + threadIdx.x;
  if (i < n) p[i] = 0u;
}
__global__ void build_H_kernel(const int* __restrict__ idx,
                               _Float16* __restrict__ Hn) {
  int t = blockIdx.x * blockDim.x + threadIdx.x; // 0..32767
  int e = t >> 3;
  int r = idx[t];
  Hn[(long)r * NODES + e] = (_Float16)0.125f;
}

__global__ void cvt_f32_f16_kernel(const float* __restrict__ s,
                                   _Float16* __restrict__ d, int n) {
  int i = blockIdx.x * blockDim.x + threadIdx.x;
  if (i < n) d[i] = (_Float16)s[i];
}

// ---------------------------------------------------------------------------
// 5) WMMA GEMM with TDM-staged, double-buffered LDS tiles.
//    C = op(A)*B (+bias)(ReLU)(+resid); 8 waves -> 32(M) x 64(N) block tile.
//    A tile:  normal  -> LDS As[m][k], 32x32, row stride 40 halves (TDM pad)
//             TRANSA  -> LDS As[k][m], 32x32, row stride 40 halves
//    B tile:  LDS Bs[k][n], 32x64, row stride 72 halves (TDM pad)
//    Wave 0 issues tensor_load_to_lds for buffer i+1, waits TENSORcnt, then
//    the block barrier publishes the tile; WMMA runs out of LDS.
// ---------------------------------------------------------------------------
static constexpr unsigned TILE_A_HALVES = 32 * 40;          // 1280
static constexpr unsigned TILE_B_HALVES = 32 * 72;          // 2304
static constexpr unsigned TILE_BUF_HALVES = TILE_A_HALVES + TILE_B_HALVES; // 3584
static constexpr unsigned TILE_BUF_BYTES  = TILE_BUF_HALVES * 2;           // 7168

template <bool TRANSA, bool BIAS, bool RELU, bool RESID, bool OUTF32>
__global__ void wmma_gemm_kernel(const _Float16* __restrict__ A,
                                 const _Float16* __restrict__ Bm,
                                 void* __restrict__ Cm,
                                 const float* __restrict__ bias,
                                 const _Float16* __restrict__ resid,
                                 int Mdim, int Ndim, int Kdim,
                                 int lda, int ldb, int ldc,
                                 long sA, long sB, long sC, long sR) {
  __shared__ _Float16 smem[2 * TILE_BUF_HALVES];

  int tid  = threadIdx.x;
  int wave = tid >> 5;
  int lane = tid & 31;
  int hi   = lane >> 4;
  int lm   = lane & 15;
  int wm   = wave >> 2;  // 0..1
  int wn   = wave & 3;   // 0..3
  int mbase = blockIdx.x * 32;
  int nbase = blockIdx.y * 64;
  int bz    = blockIdx.z;
  const _Float16* Ab = A + (long)bz * sA;
  const _Float16* Bb = Bm + (long)bz * sB;

  auto stage = [&](int buf, int k0) {
    // A tile: rows of 32 halves (16 DW) + 4 DW pad -> stride 40 halves
    const _Float16* ga = TRANSA ? (Ab + (long)k0 * lda + mbase)
                                : (Ab + (long)mbase * lda + k0);
    tdm_load_2d_f16(buf * TILE_BUF_BYTES, ga, 32, 32, lda,
                    /*interval 16DW*/ 3, /*pad 4DW*/ 3);
    // B tile: rows of 64 halves (32 DW) + 4 DW pad -> stride 72 halves
    tdm_load_2d_f16(buf * TILE_BUF_BYTES + TILE_A_HALVES * 2,
                    Bb + (long)k0 * ldb + nbase, 64, 32, ldb,
                    /*interval 32DW*/ 4, /*pad 4DW*/ 3);
  };

  v8f acc;
#pragma unroll
  for (int i = 0; i < 8; i++) acc[i] = 0.0f;

  int nk = Kdim >> 5;
  if (wave == 0) stage(0, 0);

  for (int i = 0; i < nk; i++) {
    if (wave == 0) {
      if (i + 1 < nk) {
        stage((i + 1) & 1, (i + 1) << 5);
        __builtin_amdgcn_s_wait_tensorcnt(2);  // first pair (buf i) complete
      } else {
        __builtin_amdgcn_s_wait_tensorcnt(0);
      }
    }
    __syncthreads();  // publish TDM-written tile to all waves

    const _Float16* As = smem + (unsigned)(i & 1) * TILE_BUF_HALVES;
    const _Float16* Bs = As + TILE_A_HALVES;

    v16h a, bf;
#pragma unroll
    for (int j = 0; j < 8; j++) {
      int ka = (j < 4 ? 2 * j : 2 * j + 8) + hi * 8;
      if (TRANSA) {
        a[2 * j]     = As[(ka) * 40 + (wm * 16 + lm)];
        a[2 * j + 1] = As[(ka + 1) * 40 + (wm * 16 + lm)];
      } else {
        a[2 * j]     = As[(wm * 16 + lm) * 40 + ka];
        a[2 * j + 1] = As[(wm * 16 + lm) * 40 + ka + 1];
      }
      int kb = (j < 4 ? 2 * j : 2 * j + 8) + hi * 8;
      bf[2 * j]     = Bs[(kb) * 72 + (wn * 16 + lm)];
      bf[2 * j + 1] = Bs[(kb + 1) * 72 + (wn * 16 + lm)];
    }
    acc = __builtin_amdgcn_wmma_f32_16x16x32_f16(
        false, a, false, bf, (short)0, acc, false, false);

    __syncthreads();  // all reads done before this buffer is overwritten
  }

  int col = nbase + wn * 16 + lm;
  float bv = BIAS ? bias[col] : 0.0f;
#pragma unroll
  for (int v = 0; v < 8; v++) {
    int row = mbase + wm * 16 + v + 8 * hi;
    float val = acc[v] + bv;
    if (RELU) val = fmaxf(val, 0.0f);
    if (RESID) val += (float)resid[(long)bz * sR + (long)row * ldc + col];
    long ci = (long)bz * sC + (long)row * ldc + col;
    if (OUTF32) ((float*)Cm)[ci] = val;
    else        ((_Float16*)Cm)[ci] = (_Float16)val;
  }
}

// ---------------------------------------------------------------------------
// 6) BatchNorm: per-channel batch stats, then apply + global residual.
// ---------------------------------------------------------------------------
__global__ void bn_stats_kernel(const float* __restrict__ p,
                                float* __restrict__ mean,
                                float* __restrict__ rstd) {
  int c = blockIdx.x;
  float s = 0.f, s2 = 0.f;
  for (int m = threadIdx.x; m < ROWS; m += 256) {
    float v = p[(long)m * CH + c];
    s += v; s2 += v * v;
  }
  __shared__ float sh[256], sh2[256];
  sh[threadIdx.x] = s; sh2[threadIdx.x] = s2;
  __syncthreads();
  for (int o = 128; o > 0; o >>= 1) {
    if (threadIdx.x < o) {
      sh[threadIdx.x] += sh[threadIdx.x + o];
      sh2[threadIdx.x] += sh2[threadIdx.x + o];
    }
    __syncthreads();
  }
  if (threadIdx.x == 0) {
    float mu = sh[0] / (float)ROWS;
    float var = sh2[0] / (float)ROWS - mu * mu;
    mean[c] = mu;
    rstd[c] = rsqrtf(var + 1e-5f);
  }
}

__global__ void bn_apply_kernel(const float* __restrict__ feat,
                                const float* __restrict__ p,
                                const float* __restrict__ mean,
                                const float* __restrict__ rstd,
                                const float* __restrict__ gamma,
                                const float* __restrict__ beta,
                                float* __restrict__ out) {
  long i = (long)blockIdx.x * blockDim.x + threadIdx.x;
  if (i >= (long)BATCH * CH * NODES) return;
  int b = (int)(i >> 20);        // C*N = 2^20
  int c = (int)((i >> 12) & 255);
  int n = (int)(i & 4095);
  float pv = p[(((long)b << 12) + n) * CH + c];
  out[i] = feat[i] + gamma[c] * ((pv - mean[c]) * rstd[c]) + beta[c];
}

// ---------------------------------------------------------------------------
// Host launch
// ---------------------------------------------------------------------------
extern "C" void kernel_launch(void* const* d_in, const int* in_sizes, int n_in,
                              void* d_out, int out_size, void* d_ws, size_t ws_size,
                              hipStream_t stream) {
  (void)in_sizes; (void)n_in; (void)out_size; (void)ws_size;
  const float* feat    = (const float*)d_in[0];
  const float* th0w    = (const float*)d_in[1];
  const float* th0b    = (const float*)d_in[2];
  const float* ph0w    = (const float*)d_in[3];
  const float* ph0b    = (const float*)d_in[4];
  const float* th1w    = (const float*)d_in[5];
  const float* th1b    = (const float*)d_in[6];
  const float* ph1w    = (const float*)d_in[7];
  const float* ph1b    = (const float*)d_in[8];
  const float* pjw     = (const float*)d_in[9];
  const float* pjb     = (const float*)d_in[10];
  const float* gamma   = (const float*)d_in[11];
  const float* beta    = (const float*)d_in[12];
  float* out = (float*)d_out;

  // Workspace carve-up (all offsets 256B-aligned).
  char* w = (char*)d_ws;
  size_t off = 0;
  auto alloc = [&](size_t bytes) { void* p = w + off; off += (bytes + 255) & ~(size_t)255; return p; };
  _Float16* Xh  = (_Float16*)alloc((size_t)ROWS * CH * 2);        // 8 MB (reused as y later)
  _Float16* f0n = (_Float16*)alloc((size_t)NODES * CH * 2);       // 2 MB
  int*      idx = (int*)alloc((size_t)NODES * KTOP * 4);          // 128 KB
  _Float16* Hn  = (_Float16*)alloc((size_t)NODES * NODES * 2);    // 32 MB
  _Float16* th0h = (_Float16*)alloc((size_t)CH * CH * 2);
  _Float16* ph0h = (_Float16*)alloc((size_t)CH * CH * 2);
  _Float16* th1h = (_Float16*)alloc((size_t)CH * CH * 2);
  _Float16* ph1h = (_Float16*)alloc((size_t)CH * CH * 2);
  _Float16* pjh  = (_Float16*)alloc((size_t)CH * CH * 2);
  _Float16* buf1 = (_Float16*)alloc((size_t)ROWS * CH * 2);       // xt
  _Float16* buf2 = (_Float16*)alloc((size_t)ROWS * CH * 2);       // ef
  _Float16* buf3 = (_Float16*)alloc((size_t)ROWS * CH * 2);       // xp
  _Float16* x1   = (_Float16*)alloc((size_t)ROWS * CH * 2);       // layer-0 out
  float*    pf   = (float*)alloc((size_t)ROWS * CH * 4);          // proj out f32
  float*    mean = (float*)alloc((size_t)CH * 4);
  float*    rstd = (float*)alloc((size_t)CH * 4);
  _Float16* y = Xh; // reuse: X only needed for theta0 GEMM

  const long sN = (long)NODES * CH; // per-batch activation stride

  // 1) transpose + cast
  transpose_cast_kernel<<<dim3(CH / 32, NODES / 32, BATCH), dim3(32, 8), 0, stream>>>(feat, Xh);
  // 2) normalize batch-0 rows
  normalize_f0_kernel<<<NODES / 256, 256, 0, stream>>>(feat, f0n);
  // 3) fused similarity + top-8
  sim_topk_kernel<<<NODES / 16, 256, 0, stream>>>(f0n, idx);
  // 4) build Hn
  {
    long nwords = (long)NODES * NODES / 2;
    zero_u32_kernel<<<(unsigned)((nwords + 255) / 256), 256, 0, stream>>>((unsigned*)Hn, nwords);
    build_H_kernel<<<(NODES * KTOP) / 256, 256, 0, stream>>>(idx, Hn);
  }
  // 5) convert weights to f16
  cvt_f32_f16_kernel<<<CH * CH / 256, 256, 0, stream>>>(th0w, th0h, CH * CH);
  cvt_f32_f16_kernel<<<CH * CH / 256, 256, 0, stream>>>(ph0w, ph0h, CH * CH);
  cvt_f32_f16_kernel<<<CH * CH / 256, 256, 0, stream>>>(th1w, th1h, CH * CH);
  cvt_f32_f16_kernel<<<CH * CH / 256, 256, 0, stream>>>(ph1w, ph1h, CH * CH);
  cvt_f32_f16_kernel<<<CH * CH / 256, 256, 0, stream>>>(pjw, pjh, CH * CH);

  dim3 gAct(ROWS / 32, CH / 64, 1);        // activation GEMMs, batch folded into M
  dim3 gHyp(NODES / 32, CH / 64, BATCH);   // hypergraph GEMMs, batched

  // Layer 0: xt = X @ th0 + b
  wmma_gemm_kernel<false, true, false, false, false><<<gAct, 256, 0, stream>>>(
      Xh, th0h, buf1, th0b, nullptr, ROWS, CH, CH, CH, CH, CH, 0, 0, 0, 0);
  // ef = H^T @ xt
  wmma_gemm_kernel<true, false, false, false, false><<<gHyp, 256, 0, stream>>>(
      Hn, buf1, buf2, nullptr, nullptr, NODES, CH, NODES, NODES, CH, CH, 0, sN, sN, 0);
  // xp = ef @ ph0 + b
  wmma_gemm_kernel<false, true, false, false, false><<<gAct, 256, 0, stream>>>(
      buf2, ph0h, buf3, ph0b, nullptr, ROWS, CH, CH, CH, CH, CH, 0, 0, 0, 0);
  // x1 = relu(H @ xp)
  wmma_gemm_kernel<false, false, true, false, false><<<gHyp, 256, 0, stream>>>(
      Hn, buf3, x1, nullptr, nullptr, NODES, CH, NODES, NODES, CH, CH, 0, sN, sN, 0);

  // Layer 1: xt2 = x1 @ th1 + b
  wmma_gemm_kernel<false, true, false, false, false><<<gAct, 256, 0, stream>>>(
      x1, th1h, buf1, th1b, nullptr, ROWS, CH, CH, CH, CH, CH, 0, 0, 0, 0);
  // ef2 = H^T @ xt2
  wmma_gemm_kernel<true, false, false, false, false><<<gHyp, 256, 0, stream>>>(
      Hn, buf1, buf2, nullptr, nullptr, NODES, CH, NODES, NODES, CH, CH, 0, sN, sN, 0);
  // xp2 = ef2 @ ph1 + b
  wmma_gemm_kernel<false, true, false, false, false><<<gAct, 256, 0, stream>>>(
      buf2, ph1h, buf3, ph1b, nullptr, ROWS, CH, CH, CH, CH, CH, 0, 0, 0, 0);
  // y = relu(H @ xp2) + x1
  wmma_gemm_kernel<false, false, true, true, false><<<gHyp, 256, 0, stream>>>(
      Hn, buf3, y, nullptr, x1, NODES, CH, NODES, NODES, CH, CH, 0, sN, sN, sN);

  // proj (f32 out): p = y @ pjw + pjb
  wmma_gemm_kernel<false, true, false, false, true><<<gAct, 256, 0, stream>>>(
      y, pjh, pf, pjb, nullptr, ROWS, CH, CH, CH, CH, CH, 0, 0, 0, 0);

  // BatchNorm + residual
  bn_stats_kernel<<<CH, 256, 0, stream>>>(pf, mean, rstd);
  bn_apply_kernel<<<(BATCH * CH * NODES) / 256, 256, 0, stream>>>(
      feat, pf, mean, rstd, gamma, beta, out);
}
